// HybridSelfAttentionBlock_65506841198824
// MI455X (gfx1250) — matrix operand order repaired and checked by reading, not compiled, not run on hardware
//
#include <hip/hip_runtime.h>
#include <hip/hip_bf16.h>
#include <math.h>

// Problem constants (match reference)
#define Bc     2
#define Lc     2048
#define Gc     128
#define LTc    2176
#define DD     512
#define PEc    64
#define DPE    576     // D + PE
#define Hh     8
#define DHh    64
#define INNERc 512
#define FFc    2048
#define Kc     32
#define SCALEc 0.125f  // DH^-0.5

typedef __attribute__((ext_vector_type(16))) _Float16 v16h;
typedef __attribute__((ext_vector_type(8)))  _Float16 v8h;
typedef __attribute__((ext_vector_type(8)))  float    v8f;

// ---------------------------------------------------------------------------
// WMMA fragment helpers (CDNA5 16-bit A 16x32 / B 32x16 lane layouts)
// A: lane<16 -> M=lane, kbase=0 ; lane>=16 -> M=lane-16, kbase=8
//    element e -> K = kbase + (e<8 ? e : 16+(e-8))   => two contiguous 8-half loads
// B: lane group gives kbase 0/8, N = lane&15, element e -> K = kbase + (e<8 ? e : 16+(e-8))
//    => two contiguous 8-half loads (same order as A)
// ---------------------------------------------------------------------------
__device__ __forceinline__ v16h load_a_frag(const _Float16* __restrict__ row, int kb, int lane) {
    const int kbase = (lane & 16) ? 8 : 0;
    v8h lo = *(const v8h*)(row + kb + kbase);
    v8h hi = *(const v8h*)(row + kb + 16 + kbase);
    v16h r;
#pragma unroll
    for (int i = 0; i < 8; ++i) { r[i] = lo[i]; r[8 + i] = hi[i]; }
    return r;
}

__device__ __forceinline__ v16h load_a_frag_f32(const float* __restrict__ row, int kb, int lane) {
    const int kbase = (lane & 16) ? 8 : 0;
    v16h r;
#pragma unroll
    for (int i = 0; i < 8; ++i) {
        r[i]     = (_Float16)row[kb + kbase + i];
        r[8 + i] = (_Float16)row[kb + kbase + 16 + i];
    }
    return r;
}

__device__ __forceinline__ v16h load_b_frag(const _Float16* __restrict__ btrow, int kb, int lane) {
    const int kbase = (lane & 16) ? 8 : 0;
    v8h lo = *(const v8h*)(btrow + kb + kbase);
    v8h hi = *(const v8h*)(btrow + kb + 16 + kbase);
    v16h r;
#pragma unroll
    for (int i = 0; i < 8; ++i) { r[i] = lo[i]; r[8 + i] = hi[i]; }
    return r;
}

__device__ __forceinline__ v8f wmma16(v16h a, v16h b, v8f c) {
    return __builtin_amdgcn_wmma_f32_16x16x32_f16(false, a, false, b, (short)0, c, false, false);
}

__device__ __forceinline__ float gelu_exact(float v) {
    return 0.5f * v * (1.0f + erff(v * 0.70710678118654752f));
}

// ---------------------------------------------------------------------------
// Generic WMMA GEMM:  C[M,N] = A_f16[M,Kd] * Bt_f16[N,Kd]^T  (+bias)(gelu)(+res)
// grid = (N/64, M/128), block = 256 (8 waves). Each wave: 16Mx64N register
// tile (4 f32 accumulators), double-buffered fragment pipeline so the 4
// back-to-back WMMAs overlap the next k-step's global loads.
// ---------------------------------------------------------------------------
__global__ void __launch_bounds__(256) gemm_tn(
    const _Float16* __restrict__ A, int lda,
    const _Float16* __restrict__ Bt, int Kd,
    const float* __restrict__ bias,
    const float* __restrict__ residual,
    float* __restrict__ Cf,
    _Float16* __restrict__ Ch,
    int N, int gelu_flag)
{
    const int lane = threadIdx.x & 31;
    const int wave = threadIdx.x >> 5;
    const int m0 = blockIdx.y * 128 + wave * 16;
    const int n0 = blockIdx.x * 64;
    const _Float16* Arow = A + (size_t)(m0 + (lane & 15)) * lda;
    const _Float16* Br0 = Bt + (size_t)(n0 +  0 + (lane & 15)) * Kd;
    const _Float16* Br1 = Bt + (size_t)(n0 + 16 + (lane & 15)) * Kd;
    const _Float16* Br2 = Bt + (size_t)(n0 + 32 + (lane & 15)) * Kd;
    const _Float16* Br3 = Bt + (size_t)(n0 + 48 + (lane & 15)) * Kd;

    v8f acc0 = {}, acc1 = {}, acc2 = {}, acc3 = {};
    v16h a  = load_a_frag(Arow, 0, lane);
    v16h b0 = load_b_frag(Br0, 0, lane);
    v16h b1 = load_b_frag(Br1, 0, lane);
    v16h b2 = load_b_frag(Br2, 0, lane);
    v16h b3 = load_b_frag(Br3, 0, lane);

    for (int kb = 32; kb < Kd; kb += 32) {
        if (kb + 224 < Kd) __builtin_prefetch(Arow + kb + 224, 0, 3);
        v16h an  = load_a_frag(Arow, kb, lane);
        v16h bn0 = load_b_frag(Br0, kb, lane);
        v16h bn1 = load_b_frag(Br1, kb, lane);
        v16h bn2 = load_b_frag(Br2, kb, lane);
        v16h bn3 = load_b_frag(Br3, kb, lane);
        acc0 = wmma16(a, b0, acc0);
        acc1 = wmma16(a, b1, acc1);
        acc2 = wmma16(a, b2, acc2);
        acc3 = wmma16(a, b3, acc3);
        a = an; b0 = bn0; b1 = bn1; b2 = bn2; b3 = bn3;
    }
    acc0 = wmma16(a, b0, acc0);
    acc1 = wmma16(a, b1, acc1);
    acc2 = wmma16(a, b2, acc2);
    acc3 = wmma16(a, b3, acc3);

    const int rb = (lane & 16) ? 8 : 0;
    v8f accs[4] = {acc0, acc1, acc2, acc3};
#pragma unroll
    for (int t = 0; t < 4; ++t) {
        const int bn = n0 + t * 16 + (lane & 15);
#pragma unroll
        for (int r = 0; r < 8; ++r) {
            const int m = m0 + rb + r;
            const size_t ci = (size_t)m * N + bn;
            float v = accs[t][r];
            if (bias) v += bias[bn];
            if (gelu_flag) v = gelu_exact(v);
            if (residual) v += residual[ci];
            if (Cf) Cf[ci] = v;
            if (Ch) Ch[ci] = (_Float16)v;
        }
    }
}

// ---------------------------------------------------------------------------
// LayerNorm over D=512, f32 in -> f16 out. One block (256 thr) per row.
// ---------------------------------------------------------------------------
__global__ void __launch_bounds__(256) layernorm_f16(
    const float* __restrict__ x, const float* __restrict__ g,
    const float* __restrict__ bb, _Float16* __restrict__ y)
{
    __shared__ float s1[256];
    __shared__ float s2[256];
    const int row = blockIdx.x, t = threadIdx.x;
    const float* xr = x + (size_t)row * DD;
    const float a0 = xr[t], a1 = xr[t + 256];
    s1[t] = a0 + a1;
    s2[t] = a0 * a0 + a1 * a1;
    __syncthreads();
    for (int off = 128; off > 0; off >>= 1) {
        if (t < off) { s1[t] += s1[t + off]; s2[t] += s2[t + off]; }
        __syncthreads();
    }
    const float mean = s1[0] * (1.0f / DD);
    const float var  = s2[0] * (1.0f / DD) - mean * mean;
    const float rs   = rsqrtf(var + 1e-5f);
    _Float16* yr = y + (size_t)row * DD;
    yr[t]       = (_Float16)((a0 - mean) * rs * g[t] + bb[t]);
    yr[t + 256] = (_Float16)((a1 - mean) * rs * g[t + 256] + bb[t + 256]);
}

// ---------------------------------------------------------------------------
// Fused local attention: one block per (b,l).
// gather(topk) + concat(rpe) -> K/V projection (pipelined WMMA, Kd=576) into
// LDS, then 8 waves = 8 heads do softmax(q.k*scale - d^2/2s^2) @ V.
// ---------------------------------------------------------------------------
__global__ void __launch_bounds__(256) local_attn(
    const _Float16* __restrict__ xnh, const float* __restrict__ rpe,
    const float* __restrict__ dist, const int* __restrict__ topk,
    const float* __restrict__ qf, const float* __restrict__ log_sigma,
    const _Float16* __restrict__ wk_t, const _Float16* __restrict__ wv_t,
    _Float16* __restrict__ oh)
{
    __shared__ int      idx[Kc];
    __shared__ _Float16 kc[Kc * INNERc];   // 32 KB
    __shared__ _Float16 vc[Kc * INNERc];   // 32 KB
    __shared__ float    sm[Hh * Kc];

    const int bl   = blockIdx.x;           // b*L + l
    const int b    = bl / Lc;
    const int lane = threadIdx.x & 31;
    const int wave = threadIdx.x >> 5;

    if (threadIdx.x < Kc) idx[threadIdx.x] = topk[(size_t)bl * Kc + threadIdx.x];
    __syncthreads();

    const float* rbase = rpe + (size_t)bl * Kc * PEc;
    const int n0 = wave * 64;

    // --- K and V projections: ctx(32x576) @ W(576x512) -> LDS f16 ---
    for (int mat = 0; mat < 2; ++mat) {
        const _Float16* wt   = mat ? wv_t : wk_t;
        _Float16*       outc = mat ? vc   : kc;
        const _Float16* Br0 = wt + (size_t)(n0 +  0 + (lane & 15)) * DPE;
        const _Float16* Br1 = wt + (size_t)(n0 + 16 + (lane & 15)) * DPE;
        const _Float16* Br2 = wt + (size_t)(n0 + 32 + (lane & 15)) * DPE;
        const _Float16* Br3 = wt + (size_t)(n0 + 48 + (lane & 15)) * DPE;
#pragma unroll
        for (int mt = 0; mt < 2; ++mt) {
            const int m0 = mt * 16;
            const int am = m0 + (lane & 15);
            const _Float16* xrow = xnh + (size_t)(b * Lc + idx[am]) * DD;
            const float*    rrow = rbase + (size_t)am * PEc;

            v8f acc0 = {}, acc1 = {}, acc2 = {}, acc3 = {};
            v16h a  = load_a_frag(xrow, 0, lane);
            v16h b0 = load_b_frag(Br0, 0, lane);
            v16h b1 = load_b_frag(Br1, 0, lane);
            v16h b2 = load_b_frag(Br2, 0, lane);
            v16h b3 = load_b_frag(Br3, 0, lane);
#pragma unroll
            for (int kb = 32; kb < DPE; kb += 32) {
                v16h an = (kb < DD) ? load_a_frag(xrow, kb, lane)
                                    : load_a_frag_f32(rrow, kb - DD, lane);
                v16h bn0 = load_b_frag(Br0, kb, lane);
                v16h bn1 = load_b_frag(Br1, kb, lane);
                v16h bn2 = load_b_frag(Br2, kb, lane);
                v16h bn3 = load_b_frag(Br3, kb, lane);
                acc0 = wmma16(a, b0, acc0);
                acc1 = wmma16(a, b1, acc1);
                acc2 = wmma16(a, b2, acc2);
                acc3 = wmma16(a, b3, acc3);
                a = an; b0 = bn0; b1 = bn1; b2 = bn2; b3 = bn3;
            }
            acc0 = wmma16(a, b0, acc0);
            acc1 = wmma16(a, b1, acc1);
            acc2 = wmma16(a, b2, acc2);
            acc3 = wmma16(a, b3, acc3);

            const int rb = (lane & 16) ? 8 : 0;
            v8f accs[4] = {acc0, acc1, acc2, acc3};
#pragma unroll
            for (int t = 0; t < 4; ++t) {
                const int nc = n0 + t * 16 + (lane & 15);
#pragma unroll
                for (int r = 0; r < 8; ++r)
                    outc[(m0 + rb + r) * INNERc + nc] = (_Float16)accs[t][r];
            }
        }
    }
    __syncthreads();

    // --- attention: wave == head, lane == neighbor ---
    const int h = wave;
    const int j = lane;
    const float* q = qf + (size_t)bl * INNERc + h * DHh;
    float s = 0.f;
#pragma unroll
    for (int d = 0; d < DHh; ++d) s += q[d] * (float)kc[j * INNERc + h * DHh + d];
    const float dj   = dist[(size_t)bl * Kc + j];
    const float sig2 = expf(2.0f * log_sigma[h]);
    s = s * SCALEc - dj * dj / (2.0f * sig2);

    float mx = s;
    for (int off = 16; off; off >>= 1) mx = fmaxf(mx, __shfl_xor(mx, off, 32));
    const float e = expf(s - mx);
    float sum = e;
    for (int off = 16; off; off >>= 1) sum += __shfl_xor(sum, off, 32);
    sm[h * Kc + j] = e / sum;
    __syncthreads();

#pragma unroll
    for (int dd = 0; dd < 2; ++dd) {
        const int d = lane * 2 + dd;
        float o = 0.f;
#pragma unroll
        for (int jj = 0; jj < Kc; ++jj)
            o += sm[h * Kc + jj] * (float)vc[jj * INNERc + h * DHh + d];
        oh[(size_t)bl * INNERc + h * DHh + d] = (_Float16)o;
    }
}

// ---------------------------------------------------------------------------
// Cross attention: one block per (b, query). 8 waves = 8 heads.
// dynamic LDS: Hh * Nk floats (scores).
// ---------------------------------------------------------------------------
__global__ void __launch_bounds__(256) cross_attn(
    const float* __restrict__ qf, const _Float16* __restrict__ kh,
    const _Float16* __restrict__ vh, _Float16* __restrict__ oh,
    int Nq, int Nk)
{
    extern __shared__ float sc[];
    const int bq   = blockIdx.x;
    const int b    = bq / Nq;
    const int lane = threadIdx.x & 31;
    const int h    = threadIdx.x >> 5;
    const float* q = qf + (size_t)bq * INNERc + h * DHh;

    float mx = -3.4e38f;
    for (int j = lane; j < Nk; j += 32) {
        const _Float16* krow = kh + (size_t)(b * Nk + j) * INNERc + h * DHh;
        float s = 0.f;
#pragma unroll
        for (int d = 0; d < DHh; ++d) s += q[d] * (float)krow[d];
        s *= SCALEc;
        sc[h * Nk + j] = s;
        mx = fmaxf(mx, s);
    }
    for (int off = 16; off; off >>= 1) mx = fmaxf(mx, __shfl_xor(mx, off, 32));
    __syncthreads();

    float sum = 0.f;
    for (int j = lane; j < Nk; j += 32) {
        const float e = expf(sc[h * Nk + j] - mx);
        sc[h * Nk + j] = e;
        sum += e;
    }
    for (int off = 16; off; off >>= 1) sum += __shfl_xor(sum, off, 32);
    __syncthreads();

    const float inv = 1.0f / sum;
#pragma unroll
    for (int dd = 0; dd < 2; ++dd) {
        const int d = lane * 2 + dd;
        float o = 0.f;
        for (int j = 0; j < Nk; ++j)
            o += sc[h * Nk + j] * (float)vh[(size_t)(b * Nk + j) * INNERc + h * DHh + d];
        oh[(size_t)bq * INNERc + h * DHh + d] = (_Float16)(o * inv);
    }
}

// ---------------------------------------------------------------------------
// Elementwise utility kernels
// ---------------------------------------------------------------------------
__global__ void split_latents(const float* __restrict__ lat,
                              float* __restrict__ spatial, float* __restrict__ glob)
{
    const size_t n = (size_t)Bc * LTc * DD;
    for (size_t i = (size_t)blockIdx.x * blockDim.x + threadIdx.x; i < n;
         i += (size_t)gridDim.x * blockDim.x) {
        const int d = (int)(i % DD);
        const int r = (int)((i / DD) % LTc);
        const int b = (int)(i / ((size_t)DD * LTc));
        const float v = lat[i];
        if (r < Lc) spatial[(size_t)(b * Lc + r) * DD + d] = v;
        else        glob[(size_t)(b * Gc + (r - Lc)) * DD + d] = v;
    }
}

__global__ void concat_out(const float* __restrict__ spatial,
                           const float* __restrict__ glob, float* __restrict__ out)
{
    const size_t n = (size_t)Bc * LTc * DD;
    for (size_t i = (size_t)blockIdx.x * blockDim.x + threadIdx.x; i < n;
         i += (size_t)gridDim.x * blockDim.x) {
        const int d = (int)(i % DD);
        const int r = (int)((i / DD) % LTc);
        const int b = (int)(i / ((size_t)DD * LTc));
        out[i] = (r < Lc) ? spatial[(size_t)(b * Lc + r) * DD + d]
                          : glob[(size_t)(b * Gc + (r - Lc)) * DD + d];
    }
}

__global__ void cvt_f32_f16(const float* __restrict__ x, _Float16* __restrict__ y, size_t n)
{
    for (size_t i = (size_t)blockIdx.x * blockDim.x + threadIdx.x; i < n;
         i += (size_t)gridDim.x * blockDim.x)
        y[i] = (_Float16)x[i];
}

// W (Kd x N) f32 row-major  ->  Wt (N x Kd) f16 row-major
__global__ void transpose_w(const float* __restrict__ W, _Float16* __restrict__ Wt,
                            int Kd, int N)
{
    const size_t n = (size_t)Kd * N;
    for (size_t i = (size_t)blockIdx.x * blockDim.x + threadIdx.x; i < n;
         i += (size_t)gridDim.x * blockDim.x) {
        const int k = (int)(i / N);
        const int c = (int)(i % N);
        Wt[(size_t)c * Kd + k] = (_Float16)W[i];
    }
}

// ---------------------------------------------------------------------------
// Host orchestration
// ---------------------------------------------------------------------------
extern "C" void kernel_launch(void* const* d_in, const int* in_sizes, int n_in,
                              void* d_out, int out_size, void* d_ws, size_t ws_size,
                              hipStream_t stream)
{
    (void)in_sizes; (void)n_in; (void)out_size; (void)ws_size;

    const float* latents   = (const float*)d_in[0];
    const float* rpe       = (const float*)d_in[1];
    const float* distances = (const float*)d_in[2];
    const float* sn_g      = (const float*)d_in[3];
    const float* sn_b      = (const float*)d_in[4];
    const float* la_wq     = (const float*)d_in[5];
    const float* la_wk     = (const float*)d_in[6];
    const float* la_wv     = (const float*)d_in[7];
    const float* la_wo     = (const float*)d_in[8];
    const float* la_bo     = (const float*)d_in[9];
    const float* log_sigma = (const float*)d_in[10];
    const float* scn_g     = (const float*)d_in[11];
    const float* scn_b     = (const float*)d_in[12];
    const float* s2g_wq    = (const float*)d_in[13];
    const float* s2g_wk    = (const float*)d_in[14];
    const float* s2g_wv    = (const float*)d_in[15];
    const float* s2g_wo    = (const float*)d_in[16];
    const float* s2g_bo    = (const float*)d_in[17];
    const float* gcn_g     = (const float*)d_in[18];
    const float* gcn_b     = (const float*)d_in[19];
    const float* g2s_wq    = (const float*)d_in[20];
    const float* g2s_wk    = (const float*)d_in[21];
    const float* g2s_wv    = (const float*)d_in[22];
    const float* g2s_wo    = (const float*)d_in[23];
    const float* g2s_bo    = (const float*)d_in[24];
    const float* gfn_g     = (const float*)d_in[25];
    const float* gfn_b     = (const float*)d_in[26];
    const float* gff_w1    = (const float*)d_in[27];
    const float* gff_b1    = (const float*)d_in[28];
    const float* gff_w2    = (const float*)d_in[29];
    const float* gff_b2    = (const float*)d_in[30];
    const float* sfn_g     = (const float*)d_in[31];
    const float* sfn_b     = (const float*)d_in[32];
    const float* sff_w1    = (const float*)d_in[33];
    const float* sff_b1    = (const float*)d_in[34];
    const float* sff_w2    = (const float*)d_in[35];
    const float* sff_b2    = (const float*)d_in[36];
    const int*   topk      = (const int*)d_in[37];

    // ---- workspace carve-out ----
    char*  ws  = (char*)d_ws;
    size_t off = 0;
    auto alloc = [&](size_t bytes) -> void* {
        off = (off + 255) & ~(size_t)255;
        void* p = ws + off;
        off += bytes;
        return p;
    };
    const size_t BL = (size_t)Bc * Lc, BG = (size_t)Bc * Gc;

    float*     spatial = (float*)alloc(BL * DD * 4);
    float*     glob    = (float*)alloc(BG * DD * 4);
    _Float16*  xnh     = (_Float16*)alloc(BL * DD * 2);
    _Float16*  xgh     = (_Float16*)alloc(BG * DD * 2);
    _Float16*  gbh     = (_Float16*)alloc(BG * DD * 2);
    _Float16*  sph     = (_Float16*)alloc(BL * DD * 2);
    float*     qf      = (float*)alloc(BL * INNERc * 4);
    _Float16*  kh      = (_Float16*)alloc(BL * INNERc * 2);
    _Float16*  vh      = (_Float16*)alloc(BL * INNERc * 2);
    _Float16*  oh      = (_Float16*)alloc(BL * INNERc * 2);
    _Float16*  ff1h    = (_Float16*)alloc(BL * FFc * 2);
    _Float16*  t_la_wq = (_Float16*)alloc((size_t)INNERc * DD * 2);
    _Float16*  t_la_wk = (_Float16*)alloc((size_t)INNERc * DPE * 2);
    _Float16*  t_la_wv = (_Float16*)alloc((size_t)INNERc * DPE * 2);
    _Float16*  t_la_wo = (_Float16*)alloc((size_t)DD * INNERc * 2);
    _Float16*  t_s2g_wq = (_Float16*)alloc((size_t)INNERc * DD * 2);
    _Float16*  t_s2g_wk = (_Float16*)alloc((size_t)INNERc * DD * 2);
    _Float16*  t_s2g_wv = (_Float16*)alloc((size_t)INNERc * DD * 2);
    _Float16*  t_s2g_wo = (_Float16*)alloc((size_t)DD * INNERc * 2);
    _Float16*  t_g2s_wq = (_Float16*)alloc((size_t)INNERc * DD * 2);
    _Float16*  t_g2s_wk = (_Float16*)alloc((size_t)INNERc * DD * 2);
    _Float16*  t_g2s_wv = (_Float16*)alloc((size_t)INNERc * DD * 2);
    _Float16*  t_g2s_wo = (_Float16*)alloc((size_t)DD * INNERc * 2);
    _Float16*  t_gff1   = (_Float16*)alloc((size_t)FFc * DD * 2);
    _Float16*  t_gff2   = (_Float16*)alloc((size_t)DD * FFc * 2);
    _Float16*  t_sff1   = (_Float16*)alloc((size_t)FFc * DD * 2);
    _Float16*  t_sff2   = (_Float16*)alloc((size_t)DD * FFc * 2);

    auto tp = [&](const float* W, _Float16* Wt, int Kd, int N) {
        const size_t n = (size_t)Kd * N;
        transpose_w<<<dim3((unsigned)((n + 255) / 256)), dim3(256), 0, stream>>>(W, Wt, Kd, N);
    };
    auto gemm = [&](const _Float16* A, int lda, const _Float16* Bt, int Kd,
                    const float* bias, const float* res, float* Cf, _Float16* Ch,
                    int M, int N, int gelu) {
        gemm_tn<<<dim3((unsigned)(N / 64), (unsigned)(M / 128)), dim3(256), 0, stream>>>(
            A, lda, Bt, Kd, bias, res, Cf, Ch, N, gelu);
    };

    const unsigned EW = (unsigned)(((size_t)Bc * LTc * DD + 255) / 256);

    // ---- 0: split + weight transposes ----
    split_latents<<<EW, 256, 0, stream>>>(latents, spatial, glob);
    tp(la_wq, t_la_wq, DD, INNERc);   tp(la_wk, t_la_wk, DPE, INNERc);
    tp(la_wv, t_la_wv, DPE, INNERc);  tp(la_wo, t_la_wo, INNERc, DD);
    tp(s2g_wq, t_s2g_wq, DD, INNERc); tp(s2g_wk, t_s2g_wk, DD, INNERc);
    tp(s2g_wv, t_s2g_wv, DD, INNERc); tp(s2g_wo, t_s2g_wo, INNERc, DD);
    tp(g2s_wq, t_g2s_wq, DD, INNERc); tp(g2s_wk, t_g2s_wk, DD, INNERc);
    tp(g2s_wv, t_g2s_wv, DD, INNERc); tp(g2s_wo, t_g2s_wo, INNERc, DD);
    tp(gff_w1, t_gff1, DD, FFc);      tp(gff_w2, t_gff2, FFc, DD);
    tp(sff_w1, t_sff1, DD, FFc);      tp(sff_w2, t_sff2, FFc, DD);

    // ---- 1: local neighborhood attention ----
    layernorm_f16<<<(unsigned)BL, 256, 0, stream>>>(spatial, sn_g, sn_b, xnh);
    gemm(xnh, DD, t_la_wq, DD, nullptr, nullptr, qf, nullptr, (int)BL, INNERc, 0);
    local_attn<<<(unsigned)BL, 256, 0, stream>>>(xnh, rpe, distances, topk, qf,
                                                 log_sigma, t_la_wk, t_la_wv, oh);
    gemm(oh, INNERc, t_la_wo, INNERc, la_bo, spatial, spatial, nullptr, (int)BL, DD, 0);

    // ---- 2: spatial -> glob cross attention ----
    layernorm_f16<<<(unsigned)BL, 256, 0, stream>>>(spatial, scn_g, scn_b, xnh);
    cvt_f32_f16<<<(unsigned)((BG * DD + 255) / 256), 256, 0, stream>>>(glob, gbh, BG * DD);
    gemm(xnh, DD, t_s2g_wq, DD, nullptr, nullptr, qf, nullptr, (int)BL, INNERc, 0);
    gemm(gbh, DD, t_s2g_wk, DD, nullptr, nullptr, nullptr, kh, (int)BG, INNERc, 0);
    gemm(gbh, DD, t_s2g_wv, DD, nullptr, nullptr, nullptr, vh, (int)BG, INNERc, 0);
    cross_attn<<<(unsigned)BL, 256, Hh * Gc * sizeof(float), stream>>>(qf, kh, vh, oh, Lc, Gc);
    gemm(oh, INNERc, t_s2g_wo, INNERc, s2g_bo, spatial, spatial, nullptr, (int)BL, DD, 0);

    // ---- 3: glob -> spatial cross attention (uses updated spatial) ----
    layernorm_f16<<<(unsigned)BG, 256, 0, stream>>>(glob, gcn_g, gcn_b, xgh);
    cvt_f32_f16<<<(unsigned)((BL * DD + 255) / 256), 256, 0, stream>>>(spatial, sph, BL * DD);
    gemm(xgh, DD, t_g2s_wq, DD, nullptr, nullptr, qf, nullptr, (int)BG, INNERc, 0);
    gemm(sph, DD, t_g2s_wk, DD, nullptr, nullptr, nullptr, kh, (int)BL, INNERc, 0);
    gemm(sph, DD, t_g2s_wv, DD, nullptr, nullptr, nullptr, vh, (int)BL, INNERc, 0);
    cross_attn<<<(unsigned)BG, 256, Hh * Lc * sizeof(float), stream>>>(qf, kh, vh, oh, Gc, Lc);
    gemm(oh, INNERc, t_g2s_wo, INNERc, g2s_bo, glob, glob, nullptr, (int)BG, DD, 0);

    // ---- 4: glob FFN ----
    layernorm_f16<<<(unsigned)BG, 256, 0, stream>>>(glob, gfn_g, gfn_b, xgh);
    gemm(xgh, DD, t_gff1, DD, gff_b1, nullptr, nullptr, ff1h, (int)BG, FFc, 1);
    gemm(ff1h, FFc, t_gff2, FFc, gff_b2, glob, glob, nullptr, (int)BG, DD, 0);

    // ---- 5: spatial FFN ----
    layernorm_f16<<<(unsigned)BL, 256, 0, stream>>>(spatial, sfn_g, sfn_b, xnh);
    gemm(xnh, DD, t_sff1, DD, sff_b1, nullptr, nullptr, ff1h, (int)BL, FFc, 1);
    gemm(ff1h, FFc, t_sff2, FFc, sff_b2, spatial, spatial, nullptr, (int)BL, DD, 0);

    // ---- 6: concat output ----
    concat_out<<<EW, 256, 0, stream>>>(spatial, glob, (float*)d_out);
}
